// MambaDecoderStack_80616536146524
// MI455X (gfx1250) — matrix operand order repaired and checked by reading, not compiled, not run on hardware
//
#include <hip/hip_runtime.h>
#include <hip/hip_bf16.h>
#include <math.h>

// ---------------- model constants ----------------
#define B_   2
#define L_   2048
#define DM_  1024
#define NL_  4
#define DS_  16
#define DC_  4
#define DI_  2048          // EXP * DM
#define DTR_ 64            // DM / 16
#define T_   (B_ * L_)     // 4096 tokens
#define EPS_ 1e-6f

typedef __bf16 bf16;
typedef __attribute__((ext_vector_type(16))) __bf16 v16bf;
typedef __attribute__((ext_vector_type(8)))  __bf16 v8bf;
typedef __attribute__((ext_vector_type(8)))  float  v8f;

__device__ __forceinline__ bf16 f2bf(float f) { return (bf16)f; }
__device__ __forceinline__ float sigmoidf_(float x) { return 1.0f / (1.0f + __expf(-x)); }

// ---------------- tiled transpose + fp32->bf16 convert ----------------
// src: K x N row-major fp32   ->   dst: N x K row-major bf16
// K, N multiples of 32. grid = (N/32, K/32), block = (32, 8)
__global__ __launch_bounds__(256)
void transpose_cvt_kernel(const float* __restrict__ src, bf16* __restrict__ dst,
                          int K, int N) {
  __shared__ float tile[32][33];
  int n0 = blockIdx.x * 32, k0 = blockIdx.y * 32;
  #pragma unroll
  for (int i = 0; i < 32; i += 8)
    tile[threadIdx.y + i][threadIdx.x] =
        src[(size_t)(k0 + threadIdx.y + i) * N + (n0 + threadIdx.x)];
  __syncthreads();
  #pragma unroll
  for (int i = 0; i < 32; i += 8)
    dst[(size_t)(n0 + threadIdx.y + i) * K + (k0 + threadIdx.x)] =
        f2bf(tile[threadIdx.x][threadIdx.y + i]);
}

// ---------------- RMSNorm -> bf16 ----------------
// one block (256 threads) per token; DM_ = 1024 = 4 * 256
__global__ __launch_bounds__(256)
void rmsnorm_kernel(const float* __restrict__ x, const float* __restrict__ w,
                    bf16* __restrict__ out) {
  int tok = blockIdx.x;
  const float* xr = x + (size_t)tok * DM_;
  float v[4]; float ss = 0.0f;
  #pragma unroll
  for (int i = 0; i < 4; ++i) {
    v[i] = xr[threadIdx.x + i * 256];
    ss += v[i] * v[i];
  }
  #pragma unroll
  for (int off = 16; off > 0; off >>= 1) ss += __shfl_xor(ss, off, 32);
  __shared__ float red[8];
  int wave = threadIdx.x >> 5, lane = threadIdx.x & 31;
  if (lane == 0) red[wave] = ss;
  __syncthreads();
  float tot = 0.0f;
  #pragma unroll
  for (int i = 0; i < 8; ++i) tot += red[i];
  float scale = rsqrtf(tot * (1.0f / DM_) + EPS_);
  #pragma unroll
  for (int i = 0; i < 4; ++i) {
    int c = threadIdx.x + i * 256;
    out[(size_t)tok * DM_ + c] = f2bf(v[i] * scale * w[c]);
  }
}

// ---------------- bf16 WMMA GEMM ----------------
// C[M,N] = A[M,K] (bf16 row-major) @ BT[N,K]^T (bf16, transposed weight, N-major)
// MODE 0: C = acc                      (plain fp32 out)
// MODE 1: C = softplus(acc + bias[n])  (delta path)
// MODE 2: C = acc + res[m,n]           (residual add)
// block = 256 threads = 8 waves (2 M x 4 N); block tile 64(M) x 128(N);
// wave tile 32 x 32 = 4 WMMA per 32-K step (2 A frags x 2 B frags).
// M multiple of 64, K multiple of 32. GUARD=false requires N multiple of 128.
template <bool GUARD, int MODE>
__global__ __launch_bounds__(256)
void gemm_bf16_kernel(const bf16* __restrict__ A, const bf16* __restrict__ BT,
                      float* __restrict__ C, int M, int N, int K,
                      const float* __restrict__ bias,
                      const float* __restrict__ res) {
  const int wave = threadIdx.x >> 5;
  const int lane = threadIdx.x & 31;
  const int half = lane >> 4;
  const int lm   = lane & 15;

  const int m0 = blockIdx.y * 64 + (wave >> 2) * 32;   // wave row tile (32 rows)
  const int n0 = blockIdx.x * 128 + (wave & 3) * 32;   // wave col tile (32 cols)

  const int col0 = n0 + lm;
  const int col1 = n0 + 16 + lm;
  bool ok0 = true, ok1 = true;
  if (GUARD) { ok0 = col0 < N; ok1 = col1 < N; }

  // A fragment rows; per-lane: elems 0..15 <- contiguous K[k0+16h .. +16)
  const bf16* pa0 = A + (size_t)(m0 + lm) * K + half * 16;
  const bf16* pa1 = A + (size_t)(m0 + 16 + lm) * K + half * 16;
  // B fragment: contiguous K[k0+16h .. +16) at this lane's column
  const bf16* pb0 = BT + (size_t)col0 * K + half * 16;
  const bf16* pb1 = BT + (size_t)col1 * K + half * 16;

  v8f acc00 = {}, acc01 = {}, acc10 = {}, acc11 = {};

  for (int k0 = 0; k0 < K; k0 += 32) {
    v16bf a0, a1;
    {
      v8bf lo = *(const v8bf*)(pa0);
      v8bf hi = *(const v8bf*)(pa0 + 8);
      #pragma unroll
      for (int i = 0; i < 8; ++i) { a0[i] = lo[i]; a0[8 + i] = hi[i]; }
    }
    {
      v8bf lo = *(const v8bf*)(pa1);
      v8bf hi = *(const v8bf*)(pa1 + 8);
      #pragma unroll
      for (int i = 0; i < 8; ++i) { a1[i] = lo[i]; a1[8 + i] = hi[i]; }
    }
    v16bf b0, b1;
    if (GUARD) {
      v16bf z = {};
      b0 = ok0 ? *(const v16bf*)(pb0) : z;
      b1 = ok1 ? *(const v16bf*)(pb1) : z;
    } else {
      b0 = *(const v16bf*)(pb0);
      b1 = *(const v16bf*)(pb1);
    }

    acc00 = __builtin_amdgcn_wmma_f32_16x16x32_bf16(false, a0, false, b0,
                                                    (short)0, acc00, false, false);
    acc01 = __builtin_amdgcn_wmma_f32_16x16x32_bf16(false, a0, false, b1,
                                                    (short)0, acc01, false, false);
    acc10 = __builtin_amdgcn_wmma_f32_16x16x32_bf16(false, a1, false, b0,
                                                    (short)0, acc10, false, false);
    acc11 = __builtin_amdgcn_wmma_f32_16x16x32_bf16(false, a1, false, b1,
                                                    (short)0, acc11, false, false);

    pa0 += 32; pa1 += 32; pb0 += 32; pb1 += 32;
  }

  // Epilogue. C/D layout: VGPR r -> row (tile_m + 8*half + r), col (tile_n + lm)
  #pragma unroll
  for (int r = 0; r < 8; ++r) {
    int r0 = m0 + half * 8 + r;        // acc00 / acc01 rows
    int r1 = r0 + 16;                  // acc10 / acc11 rows
    float v00 = acc00[r], v01 = acc01[r], v10 = acc10[r], v11 = acc11[r];
    if (MODE == 1) {
      if (ok0) {
        float b = bias[col0];
        float t0 = v00 + b; v00 = (t0 > 20.0f) ? t0 : log1pf(__expf(t0));
        float t1 = v10 + b; v10 = (t1 > 20.0f) ? t1 : log1pf(__expf(t1));
      }
      if (ok1) {
        float b = bias[col1];
        float t0 = v01 + b; v01 = (t0 > 20.0f) ? t0 : log1pf(__expf(t0));
        float t1 = v11 + b; v11 = (t1 > 20.0f) ? t1 : log1pf(__expf(t1));
      }
    } else if (MODE == 2) {
      if (ok0) { v00 += res[(size_t)r0 * N + col0]; v10 += res[(size_t)r1 * N + col0]; }
      if (ok1) { v01 += res[(size_t)r0 * N + col1]; v11 += res[(size_t)r1 * N + col1]; }
    }
    if (ok0) { C[(size_t)r0 * N + col0] = v00; C[(size_t)r1 * N + col0] = v10; }
    if (ok1) { C[(size_t)r0 * N + col1] = v01; C[(size_t)r1 * N + col1] = v11; }
  }
}

// ---------------- depthwise causal conv (K=4) + bias + SiLU ----------------
// u lives in xz columns [0, DI) with row stride 2*DI. Writes fp32 + bf16 copies.
__global__ __launch_bounds__(256)
void conv_silu_kernel(const float* __restrict__ xz, const float* __restrict__ cw,
                      const float* __restrict__ cb, float* __restrict__ uc,
                      bf16* __restrict__ ucb) {
  int idx = blockIdx.x * 256 + threadIdx.x;   // over T_ * DI_
  int d = idx % DI_;
  int t = (idx / DI_) % L_;
  int b = idx / (DI_ * L_);
  float w0 = cw[d * DC_ + 0], w1 = cw[d * DC_ + 1];
  float w2 = cw[d * DC_ + 2], w3 = cw[d * DC_ + 3];
  size_t rowbase = ((size_t)b * L_ + t) * (2 * DI_) + d;
  const size_t rs = 2 * DI_;
  float acc = cb[d];
  acc += xz[rowbase] * w3;
  if (t >= 1) acc += xz[rowbase - rs]     * w2;
  if (t >= 2) acc += xz[rowbase - 2 * rs] * w1;
  if (t >= 3) acc += xz[rowbase - 3 * rs] * w0;
  float s = acc * sigmoidf_(acc);
  size_t o = (size_t)(b * L_ + t) * DI_ + d;
  uc[o]  = s;
  ucb[o] = f2bf(s);
}

// ---------------- extract dt (first DTR_ cols of xdbl) -> bf16 ----------------
__global__ __launch_bounds__(256)
void extract_dt_kernel(const float* __restrict__ xdbl, bf16* __restrict__ dtb) {
  int idx = blockIdx.x * 256 + threadIdx.x;   // over T_ * DTR_
  int r = idx / DTR_, c = idx % DTR_;
  dtb[(size_t)r * DTR_ + c] = f2bf(xdbl[(size_t)r * (DTR_ + 2 * DS_) + c]);
}

// ---------------- selective scan + SiLU(z) gating -> bf16 ----------------
// one thread per (b, d) channel; 16 fp32 states in registers; L_ sequential steps
__global__ __launch_bounds__(256)
void scan_kernel(const float* __restrict__ delta, const float* __restrict__ uc,
                 const float* __restrict__ xdbl,  const float* __restrict__ xz,
                 const float* __restrict__ A_log, const float* __restrict__ Dsk,
                 bf16* __restrict__ yb) {
  int gid = blockIdx.x * 256 + threadIdx.x;   // 0 .. B_*DI_-1
  int b = gid / DI_, d = gid % DI_;

  float As[DS_];
  #pragma unroll
  for (int s = 0; s < DS_; ++s) As[s] = -__expf(A_log[d * DS_ + s]);
  float Dv = Dsk[d];
  float h[DS_];
  #pragma unroll
  for (int s = 0; s < DS_; ++s) h[s] = 0.0f;

  for (int t = 0; t < L_; ++t) {
    size_t tok = (size_t)b * L_ + t;
    float dt = delta[tok * DI_ + d];
    float u  = uc[tok * DI_ + d];

    const float4* bc = (const float4*)(xdbl + tok * (DTR_ + 2 * DS_) + DTR_);
    float Bv[DS_], Cv[DS_];
    #pragma unroll
    for (int i = 0; i < 4; ++i) {
      float4 q = bc[i];
      Bv[4 * i] = q.x; Bv[4 * i + 1] = q.y; Bv[4 * i + 2] = q.z; Bv[4 * i + 3] = q.w;
    }
    #pragma unroll
    for (int i = 0; i < 4; ++i) {
      float4 q = bc[4 + i];
      Cv[4 * i] = q.x; Cv[4 * i + 1] = q.y; Cv[4 * i + 2] = q.z; Cv[4 * i + 3] = q.w;
    }

    float du = dt * u;
    float y = 0.0f;
    #pragma unroll
    for (int s = 0; s < DS_; ++s) {
      float dA = __expf(dt * As[s]);
      h[s] = dA * h[s] + du * Bv[s];
      y += h[s] * Cv[s];
    }
    y += u * Dv;

    float z = xz[tok * (2 * DI_) + DI_ + d];
    yb[tok * DI_ + d] = f2bf(y * (z * sigmoidf_(z)));
  }
}

// ---------------- host launcher ----------------
extern "C" void kernel_launch(void* const* d_in, const int* in_sizes, int n_in,
                              void* d_out, int out_size, void* d_ws, size_t ws_size,
                              hipStream_t stream) {
  const float* x_in   = (const float*)d_in[0];
  const float* norm_w = (const float*)d_in[1];
  const float* in_w   = (const float*)d_in[2];
  const float* conv_w = (const float*)d_in[3];
  const float* conv_b = (const float*)d_in[4];
  const float* x_w    = (const float*)d_in[5];
  const float* dt_w   = (const float*)d_in[6];
  const float* dt_b   = (const float*)d_in[7];
  const float* A_log  = (const float*)d_in[8];
  const float* D_skip = (const float*)d_in[9];
  const float* out_w  = (const float*)d_in[10];
  float* outp = (float*)d_out;

  char* ws = (char*)d_ws;
  size_t off = 0;
  auto alloc = [&](size_t bytes) -> void* {
    void* p = ws + off;
    off += (bytes + 255) & ~(size_t)255;
    return p;
  };

  float* xcur  = (float*)alloc((size_t)T_ * DM_ * 4);
  bf16*  hb    = (bf16*) alloc((size_t)T_ * DM_ * 2);
  float* xz    = (float*)alloc((size_t)T_ * 2 * DI_ * 4);
  float* uc    = (float*)alloc((size_t)T_ * DI_ * 4);
  bf16*  ucb   = (bf16*) alloc((size_t)T_ * DI_ * 2);
  float* xdbl  = (float*)alloc((size_t)T_ * (DTR_ + 2 * DS_) * 4);
  bf16*  dtb   = (bf16*) alloc((size_t)T_ * DTR_ * 2);
  float* delta = (float*)alloc((size_t)T_ * DI_ * 4);
  bf16*  yb    = (bf16*) alloc((size_t)T_ * DI_ * 2);
  bf16*  wInT  = (bf16*) alloc((size_t)(2 * DI_) * DM_ * 2);
  bf16*  xwT   = (bf16*) alloc((size_t)(DTR_ + 2 * DS_) * DI_ * 2);
  bf16*  dtwT  = (bf16*) alloc((size_t)DI_ * DTR_ * 2);
  bf16*  owT   = (bf16*) alloc((size_t)DM_ * DI_ * 2);
  (void)ws_size; (void)in_sizes; (void)n_in; (void)out_size;

  const int NX = DTR_ + 2 * DS_;   // 96

  for (int l = 0; l < NL_; ++l) {
    const float* xin = (l == 0) ? x_in : xcur;
    float* xout = (l == NL_ - 1) ? outp : xcur;

    // weight transposes (fp32 KxN -> bf16 NxK)
    transpose_cvt_kernel<<<dim3((2 * DI_) / 32, DM_ / 32), dim3(32, 8), 0, stream>>>(
        in_w + (size_t)l * DM_ * 2 * DI_, wInT, DM_, 2 * DI_);
    transpose_cvt_kernel<<<dim3(NX / 32, DI_ / 32), dim3(32, 8), 0, stream>>>(
        x_w + (size_t)l * DI_ * NX, xwT, DI_, NX);
    transpose_cvt_kernel<<<dim3(DI_ / 32, DTR_ / 32), dim3(32, 8), 0, stream>>>(
        dt_w + (size_t)l * DTR_ * DI_, dtwT, DTR_, DI_);
    transpose_cvt_kernel<<<dim3(DM_ / 32, DI_ / 32), dim3(32, 8), 0, stream>>>(
        out_w + (size_t)l * DI_ * DM_, owT, DI_, DM_);

    // h = rmsnorm(x) -> bf16
    rmsnorm_kernel<<<T_, 256, 0, stream>>>(xin, norm_w + (size_t)l * DM_, hb);

    // xz = h @ in_w  (4096 x 4096, K=1024)   N % 128 == 0 -> unguarded
    gemm_bf16_kernel<false, 0><<<dim3((2 * DI_) / 128, T_ / 64), 256, 0, stream>>>(
        hb, wInT, xz, T_, 2 * DI_, DM_, nullptr, nullptr);

    // u = silu(causal_conv(u) + b)
    conv_silu_kernel<<<(T_ * DI_) / 256, 256, 0, stream>>>(
        xz, conv_w + (size_t)l * DI_ * DC_, conv_b + (size_t)l * DI_, uc, ucb);

    // xdbl = u @ x_w  (4096 x 96, K=2048)   N = 96 -> guarded
    gemm_bf16_kernel<true, 0><<<dim3((NX + 127) / 128, T_ / 64), 256, 0, stream>>>(
        ucb, xwT, xdbl, T_, NX, DI_, nullptr, nullptr);

    // dt -> bf16
    extract_dt_kernel<<<(T_ * DTR_) / 256, 256, 0, stream>>>(xdbl, dtb);

    // delta = softplus(dt @ dt_w + dt_b)  (4096 x 2048, K=64)  -> unguarded
    gemm_bf16_kernel<false, 1><<<dim3(DI_ / 128, T_ / 64), 256, 0, stream>>>(
        dtb, dtwT, delta, T_, DI_, DTR_, dt_b + (size_t)l * DI_, nullptr);

    // selective scan + gating -> y (bf16)
    scan_kernel<<<(B_ * DI_) / 256, 256, 0, stream>>>(
        delta, uc, xdbl, xz, A_log + (size_t)l * DI_ * DS_,
        D_skip + (size_t)l * DI_, yb);

    // x = y @ out_w + residual  (4096 x 1024, K=2048)  -> unguarded
    gemm_bf16_kernel<false, 2><<<dim3(DM_ / 128, T_ / 64), 256, 0, stream>>>(
        yb, owT, xout, T_, DM_, DI_, nullptr, xin);
  }
}
